// Model_Select_13426067767411
// MI455X (gfx1250) — matrix operand order repaired and checked by reading, not compiled, not run on hardware
//
#include <hip/hip_runtime.h>
#include <math.h>

// ---------------------------------------------------------------------------
// Types for CDNA5 WMMA (gfx1250, wave32)
// ---------------------------------------------------------------------------
typedef __attribute__((ext_vector_type(16))) __bf16 v16bf;
typedef __attribute__((ext_vector_type(8)))  __bf16 v8bf;
typedef __attribute__((ext_vector_type(8)))  float  v8f;

#if defined(__has_builtin)
#  if __has_builtin(__builtin_amdgcn_global_load_async_to_lds_b128)
#    define HAVE_ASYNC_B128 1
#  endif
#  if __has_builtin(__builtin_amdgcn_s_wait_asynccnt)
#    define HAVE_WAIT_ASYNC 1
#  endif
#endif

#if defined(HAVE_ASYNC_B128)
// The builtin's parameters are pointers to gcc-style int4 vectors:
//   arg0: global (AS1) source, arg1: LDS (AS3) destination, imm offset, imm cpol
typedef int v4i_vs __attribute__((vector_size(16)));
typedef __attribute__((address_space(1))) v4i_vs as1_v4i;
typedef __attribute__((address_space(3))) v4i_vs as3_v4i;
#endif

__device__ __forceinline__ void wait_asynccnt0()
{
#if defined(HAVE_ASYNC_B128)
#  if defined(HAVE_WAIT_ASYNC)
  __builtin_amdgcn_s_wait_asynccnt(0);
#  else
  asm volatile("s_wait_asynccnt 0x0" ::: "memory");
#  endif
#endif
}

#define TILE_M 64
#define TILE_N 64
#define TILE_K 32
#define LDS_STRIDE 40   // halves; 80B row stride, 16B aligned

// ---------------------------------------------------------------------------
// Tiled bf16 GEMM with WMMA:  C = A @ B (+bias) [+relu]
//   A: (M,K) row-major bf16, optional transB: B element (k,n) = Bm[n*ldb+k]
//   Batched via gridDim.z with two-level strides: z -> (z/H, z%H)
//   Writes f32 (Cf) and/or bf16 (Cb).
//   Interior tiles use async global->LDS copies (if available) or 128-bit
//   loads; edges fall back to guarded scalar staging.
// ---------------------------------------------------------------------------
__global__ __launch_bounds__(128) void gemm_bf16_kernel(
    const __bf16* __restrict__ A, const __bf16* __restrict__ Bm,
    const float* __restrict__ bias,
    float* __restrict__ Cf, __bf16* __restrict__ Cb,
    int M, int N, int K, int lda, int ldb, int ldc,
    long sAb, long sAh, long sBb, long sBh, long sCb, long sCh,
    int H, int transB, int relu)
{
  __shared__ __align__(16) __bf16 As [TILE_M * LDS_STRIDE]; // A tile (M x K)
  __shared__ __align__(16) __bf16 Bst[TILE_N * LDS_STRIDE]; // B tile transposed (N x K)

  const int z  = blockIdx.z;
  const int zb = z / H, zh = z % H;
  A  += (long)zb * sAb + (long)zh * sAh;
  Bm += (long)zb * sBb + (long)zh * sBh;
  const long coff = (long)zb * sCb + (long)zh * sCh;

  const int tid  = threadIdx.x;
  const int wv   = tid >> 5;        // wave 0..3 -> rows wv*16..wv*16+15
  const int lane = tid & 31;
  const int hf   = lane >> 4;       // lane half
  const int l16  = lane & 15;

  const int m0 = blockIdx.y * TILE_M;
  const int n0 = blockIdx.x * TILE_N;

  const bool aVec = ((lda & 7) == 0) && (m0 + TILE_M <= M);
  const bool bFull = (n0 + TILE_N <= N) && ((ldb & 7) == 0);

  v8f acc[4] = {};

  for (int k0 = 0; k0 < K; k0 += TILE_K) {
    const bool kfull = (k0 + TILE_K <= K);

    // ---- stage A tile (64x32) ----
    if (aVec && kfull) {
#pragma unroll
      for (int p = 0; p < 2; ++p) {
        int chunk = tid + p * 128;          // 256 chunks of 8 halves
        int r  = chunk >> 2;
        int cb = (chunk & 3) << 3;
        const __bf16* gp = &A[(long)(m0 + r) * lda + k0 + cb];
        __bf16* lp = &As[r * LDS_STRIDE + cb];
#if defined(HAVE_ASYNC_B128)
        __builtin_amdgcn_global_load_async_to_lds_b128(
            (as1_v4i*)gp, (as3_v4i*)lp, 0, 0);
#else
        *(v8bf*)lp = *(const v8bf*)gp;
        if (p == 0 && k0 + 2 * TILE_K <= K)
          __builtin_prefetch(gp + TILE_K, 0, 0);
#endif
      }
    } else {
      for (int i = tid; i < TILE_M * TILE_K; i += 128) {
        int r = i >> 5, c = i & 31;
        int gm = m0 + r, gk = k0 + c;
        __bf16 v = (__bf16)0.f;
        if (gm < M && gk < K) v = A[(long)gm * lda + gk];
        As[r * LDS_STRIDE + c] = v;
      }
    }

    // ---- stage B tile transposed: Bst[n][k] ----
    if (transB) {
      if (bFull && kfull) {
#pragma unroll
        for (int p = 0; p < 2; ++p) {
          int chunk = tid + p * 128;
          int n  = chunk >> 2;
          int cb = (chunk & 3) << 3;
          const __bf16* gp = &Bm[(long)(n0 + n) * ldb + k0 + cb];
          __bf16* lp = &Bst[n * LDS_STRIDE + cb];
#if defined(HAVE_ASYNC_B128)
          __builtin_amdgcn_global_load_async_to_lds_b128(
              (as1_v4i*)gp, (as3_v4i*)lp, 0, 0);
#else
          *(v8bf*)lp = *(const v8bf*)gp;
          if (p == 0 && k0 + 2 * TILE_K <= K)
            __builtin_prefetch(gp + TILE_K, 0, 0);
#endif
        }
      } else {
        for (int i = tid; i < TILE_N * TILE_K; i += 128) {
          int c = i & 31, n = i >> 5;
          int gk = k0 + c, gn = n0 + n;
          __bf16 v = (__bf16)0.f;
          if (gk < K && gn < N) v = Bm[(long)gn * ldb + gk];
          Bst[n * LDS_STRIDE + c] = v;
        }
      }
    } else {
      if (bFull && kfull) {
#pragma unroll
        for (int p = 0; p < 2; ++p) {
          int chunk = tid + p * 128;
          int kk = chunk >> 3;               // 0..31
          int nb = (chunk & 7) << 3;         // 0,8,..,56
          v8bf v = *(const v8bf*)&Bm[(long)(k0 + kk) * ldb + n0 + nb];
#pragma unroll
          for (int e = 0; e < 8; ++e)
            Bst[(nb + e) * LDS_STRIDE + kk] = v[e];
        }
      } else {
        for (int i = tid; i < TILE_N * TILE_K; i += 128) {
          int n = i & 63, c = i >> 6;
          int gk = k0 + c, gn = n0 + n;
          __bf16 v = (__bf16)0.f;
          if (gk < K && gn < N) v = Bm[(long)gk * ldb + gn];
          Bst[n * LDS_STRIDE + c] = v;
        }
      }
    }
    wait_asynccnt0();
    __syncthreads();

    // A fragment per ISA layout: lane row = wv*16+l16;
    // half 0: K {0..7, 16..23}; half 1: K {8..15, 24..31}  -> two 16B chunks
    v16bf afrag;
    const __bf16* ap = &As[(wv * 16 + l16) * LDS_STRIDE];
    ((v8bf*)&afrag)[0] = *(const v8bf*)(ap + hf * 8);
    ((v8bf*)&afrag)[1] = *(const v8bf*)(ap + 16 + hf * 8);

#pragma unroll
    for (int t = 0; t < 4; ++t) {
      // B fragment: lane column n = t*16+l16; half 0: K {0..7, 16..23}; half 1: K {8..15, 24..31}
      v16bf bfrag;
      const __bf16* bp = &Bst[(t * 16 + l16) * LDS_STRIDE + hf * 8];
      ((v8bf*)&bfrag)[0] = *(const v8bf*)(bp);
      ((v8bf*)&bfrag)[1] = *(const v8bf*)(bp + 16);
      acc[t] = __builtin_amdgcn_wmma_f32_16x16x32_bf16(
          false, afrag, false, bfrag, (short)0, acc[t], false, false);
    }
    __syncthreads();
  }

  float*  cfp = Cf ? Cf + coff : nullptr;
  __bf16* cbp = Cb ? Cb + coff : nullptr;
#pragma unroll
  for (int t = 0; t < 4; ++t) {
    int col = n0 + t * 16 + l16;
    if (col >= N) continue;
    float bvl = bias ? bias[col] : 0.f;
#pragma unroll
    for (int j = 0; j < 8; ++j) {
      int row = m0 + wv * 16 + hf * 8 + j;   // C layout: VGPR j -> M=j (+8 for hi half)
      if (row >= M) continue;
      float v = acc[t][j] + bvl;
      if (relu && v < 0.f) v = 0.f;
      long o = (long)row * ldc + col;
      if (cfp) cfp[o] = v;
      if (cbp) cbp[o] = (__bf16)v;
    }
  }
}

// ---------------------------------------------------------------------------
// f32 -> bf16 conversion (weights)
// ---------------------------------------------------------------------------
__global__ void cvt_f32_to_bf16_kernel(const float* __restrict__ src,
                                       __bf16* __restrict__ dst, long n)
{
  long i = (long)blockIdx.x * 256 + threadIdx.x;
  if (i < n) dst[i] = (__bf16)src[i];
}

// ---------------------------------------------------------------------------
// Row softmax over logits (Z,S,S) with scale; bf16 scores written with padded
// row stride SP (multiple of 8 halves) so the P@V GEMM can use 128-bit loads.
// Row 0 probabilities also written as f32 (for top-k).
// ---------------------------------------------------------------------------
__global__ __launch_bounds__(256) void softmax_kernel(
    const float* __restrict__ logits, __bf16* __restrict__ score,
    float* __restrict__ row0, int S, int SP, float scale)
{
  const int z = blockIdx.y, r = blockIdx.x, tid = threadIdx.x;
  const float* in  = logits + ((long)z * S + r) * S;
  __bf16*      out = score + (long)z * S * SP + (long)r * SP;
  __shared__ float sd[256];

  float mx = -3.0e38f;
  for (int j = tid; j < S; j += 256) mx = fmaxf(mx, in[j] * scale);
  sd[tid] = mx; __syncthreads();
  for (int s = 128; s > 0; s >>= 1) { if (tid < s) sd[tid] = fmaxf(sd[tid], sd[tid + s]); __syncthreads(); }
  mx = sd[0]; __syncthreads();

  float sum = 0.f;
  for (int j = tid; j < S; j += 256) sum += __expf(in[j] * scale - mx);
  sd[tid] = sum; __syncthreads();
  for (int s = 128; s > 0; s >>= 1) { if (tid < s) sd[tid] += sd[tid + s]; __syncthreads(); }
  float inv = 1.f / sd[0];

  for (int j = tid; j < S; j += 256) {
    float p = __expf(in[j] * scale - mx) * inv;
    out[j] = (__bf16)p;
    if (r == 0) row0[(long)z * S + j] = p;
  }
}

// ---------------------------------------------------------------------------
// y = LayerNorm(x + delta) * g + b ; writes f32 and bf16 (may alias x)
// D must be a multiple of 256 (1024 or 512 here).
// ---------------------------------------------------------------------------
__global__ __launch_bounds__(256) void ln_residual_kernel(
    const float* __restrict__ x, const float* __restrict__ delta,
    const float* __restrict__ g, const float* __restrict__ bb,
    float* __restrict__ outf, __bf16* __restrict__ outb, int D)
{
  const long row = blockIdx.x;
  const int tid = threadIdx.x;
  const float* xr = x + row * D;
  const float* dr = delta + row * D;
  const int nper = D >> 8;
  float v[4];
  __shared__ float sd[256];

  float s = 0.f;
  for (int i = 0; i < nper; ++i) { float t = xr[tid + (i << 8)] + dr[tid + (i << 8)]; v[i] = t; s += t; }
  sd[tid] = s; __syncthreads();
  for (int t = 128; t > 0; t >>= 1) { if (tid < t) sd[tid] += sd[tid + t]; __syncthreads(); }
  float mean = sd[0] / (float)D; __syncthreads();

  s = 0.f;
  for (int i = 0; i < nper; ++i) { float t = v[i] - mean; s += t * t; }
  sd[tid] = s; __syncthreads();
  for (int t = 128; t > 0; t >>= 1) { if (tid < t) sd[tid] += sd[tid + t]; __syncthreads(); }
  float rs = rsqrtf(sd[0] / (float)D + 1e-5f);

  for (int i = 0; i < nper; ++i) {
    int c = tid + (i << 8);
    float o = (v[i] - mean) * rs * g[c] + bb[c];
    outf[row * D + c] = o;
    outb[row * D + c] = (__bf16)o;
  }
}

// ---------------------------------------------------------------------------
// Top-k: equivalent to the reference's flatten/top_k/dedupe/sort pipeline.
// Token score = max over heads of s1*s2 (CLS row, cols 1..num); pick topk by
// (score desc, index asc); emit selected indices ascending.
// ---------------------------------------------------------------------------
__global__ __launch_bounds__(256) void topk_kernel(
    const float* __restrict__ s1, const float* __restrict__ s2,
    int S, int num, int topk, int* __restrict__ idx_out)
{
  const int b = blockIdx.x, tid = threadIdx.x;
  __shared__ float m[1024];
  __shared__ int   sel[1024];
  for (int t = tid; t < num; t += 256) {
    float a0 = s1[((long)b * 2 + 0) * S + 1 + t] * s2[((long)b * 2 + 0) * S + 1 + t];
    float a1 = s1[((long)b * 2 + 1) * S + 1 + t] * s2[((long)b * 2 + 1) * S + 1 + t];
    m[t] = fmaxf(a0, a1);
  }
  __syncthreads();
  for (int t = tid; t < num; t += 256) {
    float mt = m[t]; int rank = 0;
    for (int u = 0; u < num; ++u) {
      float mu = m[u];
      rank += (mu > mt) || (mu == mt && u < t);
    }
    sel[t] = (rank < topk) ? 1 : 0;
  }
  __syncthreads();
  for (int t = tid; t < num; t += 256) {
    if (sel[t]) {
      int pos = 0;
      for (int u = 0; u < t; ++u) pos += sel[u];
      idx_out[b * topk + pos] = t;
    }
  }
}

// ---------------------------------------------------------------------------
// Build / gather kernels
// ---------------------------------------------------------------------------
__global__ void build_xc_kernel(const float* __restrict__ ct, const float* __restrict__ cls,
                                float* __restrict__ xf, __bf16* __restrict__ xb)
{
  long i = (long)blockIdx.x * 256 + threadIdx.x;
  const long total = 8L * 1025 * 1024;
  if (i >= total) return;
  int d = (int)(i & 1023);
  long rs = i >> 10;
  int s = (int)(rs % 1025);
  int b = (int)(rs / 1025);
  float v = (s == 0) ? cls[d] : ct[(((long)b * 1024) + (s - 1)) * 1024 + d];
  xf[i] = v; xb[i] = (__bf16)v;
}

__global__ void gather_c_kernel(const float* __restrict__ xc, const int* __restrict__ idx,
                                float* __restrict__ xscf, __bf16* __restrict__ xscb,
                                float* __restrict__ cfeat)
{
  long i = (long)blockIdx.x * 256 + threadIdx.x;
  const long total = 8L * 129 * 1024;
  if (i >= total) return;
  int d = (int)(i & 1023);
  long rs = i >> 10;
  int r = (int)(rs % 129);
  int b = (int)(rs / 129);
  float v;
  if (r == 0) {
    v = xc[((long)b * 1025) * 1024 + d];
  } else {
    int t = idx[b * 128 + r - 1];
    v = xc[((long)b * 1025 + 1 + t) * 1024 + d];
    cfeat[(((long)b * 128) + (r - 1)) * 1024 + d] = v;
  }
  xscf[i] = v; xscb[i] = (__bf16)v;
}

__global__ void build_xf_kernel(const float* __restrict__ ft, const float* __restrict__ clsf,
                                const int* __restrict__ idxc,
                                float* __restrict__ xff, __bf16* __restrict__ xfb)
{
  long i = (long)blockIdx.x * 256 + threadIdx.x;
  const long total = 8L * 771 * 512;
  if (i >= total) return;
  int d = (int)(i & 511);
  long rs = i >> 9;
  int r = (int)(rs % 771);
  int b = (int)(rs / 771);
  float v;
  if (r == 0) {
    v = clsf[d];
  } else if (r <= 768) {               // bftp: cr[:, :, :3072].reshape(768,512)
    int q = r - 1;
    int tok = idxc[b * 128 + q / 6];
    v = ft[(((long)b * 1024) + tok) * 4096 + (q % 6) * 512 + d];
  } else {                             // va: cr[:, 0, 3072:].reshape(2,512)
    int tok = idxc[b * 128];
    v = ft[(((long)b * 1024) + tok) * 4096 + 3072 + (r - 769) * 512 + d];
  }
  xff[i] = v; xfb[i] = (__bf16)v;
}

__global__ void gather_f_kernel(const float* __restrict__ xf, const int* __restrict__ idx,
                                float* __restrict__ xsff, __bf16* __restrict__ xsfb,
                                float* __restrict__ ffeat)
{
  long i = (long)blockIdx.x * 256 + threadIdx.x;
  const long total = 8L * 257 * 512;
  if (i >= total) return;
  int d = (int)(i & 511);
  long rs = i >> 9;
  int r = (int)(rs % 257);
  int b = (int)(rs / 257);
  float v;
  if (r == 0) {
    v = xf[((long)b * 771) * 512 + d];
  } else {
    int t = idx[b * 256 + r - 1];
    v = xf[((long)b * 771 + 1 + t) * 512 + d];
    ffeat[(((long)b * 256) + (r - 1)) * 512 + d] = v;
  }
  xsff[i] = v; xsfb[i] = (__bf16)v;
}

// rel = x_row0 @ W(D,8) + bias -> (B,8)
__global__ __launch_bounds__(256) void rel_kernel(
    const float* __restrict__ x, long rowStride,
    const float* __restrict__ w, const float* __restrict__ bias,
    float* __restrict__ out, int D)
{
  const int b = blockIdx.x;
  const float* xr = x + (long)b * rowStride;
  __shared__ float sd[256];
  for (int j = 0; j < 8; ++j) {
    float s = 0.f;
    for (int k = threadIdx.x; k < D; k += 256) s += xr[k] * w[k * 8 + j];
    sd[threadIdx.x] = s; __syncthreads();
    for (int t = 128; t > 0; t >>= 1) { if (threadIdx.x < t) sd[threadIdx.x] += sd[threadIdx.x + t]; __syncthreads(); }
    if (threadIdx.x == 0) out[b * 8 + j] = sd[0] + bias[j];
    __syncthreads();
  }
}

__global__ void final_max_kernel(const float* __restrict__ rc, const float* __restrict__ rf,
                                 float* __restrict__ out)
{
  int i = threadIdx.x;
  if (i < 64) out[i] = fmaxf(rc[i], rf[i]);
}

// ---------------------------------------------------------------------------
// Host-side orchestration
// ---------------------------------------------------------------------------
struct LayerP {
  const float *wq,*bq,*wk,*bk,*wv,*bv,*wo,*bo,*ln1g,*ln1b,*w1,*b1,*w2,*b2,*ln2g,*ln2b;
};

static LayerP layer_from(void* const* d_in, int base) {
  LayerP p;
  const float** f = (const float**)&p;
  for (int i = 0; i < 16; ++i) f[i] = (const float*)d_in[base + i];
  return p;
}

static void launch_gemm(hipStream_t st,
    const __bf16* A, int lda, long sAb, long sAh,
    const __bf16* Bm, int ldb, long sBb, long sBh, int transB,
    const float* bias, float* Cf, __bf16* Cb, int ldc, long sCb, long sCh,
    int M, int N, int K, int batches, int H, int relu)
{
  dim3 grid((N + TILE_N - 1) / TILE_N, (M + TILE_M - 1) / TILE_M, batches);
  gemm_bf16_kernel<<<grid, dim3(128), 0, st>>>(A, Bm, bias, Cf, Cb, M, N, K,
      lda, ldb, ldc, sAb, sAh, sBb, sBh, sCb, sCh, H, transB, relu);
}

// One full encoder layer, in-place on (xf, xb) of shape (8, S, D)
static void run_layer(hipStream_t st, int S, int D,
    float* xf, __bf16* xb, const LayerP& P, __bf16* const* wbf,
    __bf16* qb, __bf16* kb, __bf16* vb, float* logits, __bf16* score,
    __bf16* attn, float* tmp, __bf16* hid, float* srow)
{
  const int Bn = 8, H = 2;
  const int M = Bn * S, dh = D / 2;
  const int SP = (S + 7) & ~7;          // padded score row stride (halves)

  launch_gemm(st, xb, D, 0, 0, wbf[0], D, 0, 0, 0, P.bq, nullptr, qb, D, 0, 0, M, D, D, 1, 1, 0);
  launch_gemm(st, xb, D, 0, 0, wbf[1], D, 0, 0, 0, P.bk, nullptr, kb, D, 0, 0, M, D, D, 1, 1, 0);
  launch_gemm(st, xb, D, 0, 0, wbf[2], D, 0, 0, 0, P.bv, nullptr, vb, D, 0, 0, M, D, D, 1, 1, 0);

  // logits[z] = Q_z @ K_z^T   (z = b*2 + h)
  launch_gemm(st, qb, D, (long)S * D, dh,
                  kb, D, (long)S * D, dh, 1,
              nullptr, logits, nullptr, S, 2L * S * S, (long)S * S,
              S, S, dh, Bn * H, H, 0);
  softmax_kernel<<<dim3(S, Bn * H), dim3(256), 0, st>>>(logits, score, srow, S, SP,
                                                        1.0f / sqrtf((float)dh));
  // attn[z] = P_z @ V_z   (scores at padded stride SP)
  launch_gemm(st, score, SP, 2L * S * SP, (long)S * SP,
                  vb, D, (long)S * D, dh, 0,
              nullptr, nullptr, attn, D, (long)S * D, dh,
              S, dh, S, Bn * H, H, 0);

  launch_gemm(st, attn, D, 0, 0, wbf[3], D, 0, 0, 0, P.bo, tmp, nullptr, D, 0, 0, M, D, D, 1, 1, 0);
  ln_residual_kernel<<<dim3(M), dim3(256), 0, st>>>(xf, tmp, P.ln1g, P.ln1b, xf, xb, D);

  launch_gemm(st, xb, D, 0, 0, wbf[4], D, 0, 0, 0, P.b1, nullptr, hid, D, 0, 0, M, D, D, 1, 1, 1);
  launch_gemm(st, hid, D, 0, 0, wbf[5], D, 0, 0, 0, P.b2, tmp, nullptr, D, 0, 0, M, D, D, 1, 1, 0);
  ln_residual_kernel<<<dim3(M), dim3(256), 0, st>>>(xf, tmp, P.ln2g, P.ln2b, xf, xb, D);
}

extern "C" void kernel_launch(void* const* d_in, const int* in_sizes, int n_in,
                              void* d_out, int out_size, void* d_ws, size_t ws_size,
                              hipStream_t stream)
{
  (void)in_sizes; (void)n_in; (void)out_size; (void)ws_size;

  // Flattened inputs (setup_inputs() dict insertion order, recursive):
  const float* c_tokens = (const float*)d_in[0];   // (8,1024,1024)
  const float* f_tokens = (const float*)d_in[1];   // (8,1024,4096)
  const float* cls_c    = (const float*)d_in[2];   // (1024)
  const float* cls_f    = (const float*)d_in[3];   // (512)
  LayerP Lca1 = layer_from(d_in, 4);
  LayerP Lca2 = layer_from(d_in, 20);
  LayerP Lcac = layer_from(d_in, 36);
  LayerP Lfa1 = layer_from(d_in, 52);
  LayerP Lfa2 = layer_from(d_in, 68);
  LayerP Lfac = layer_from(d_in, 84);
  const float* wc  = (const float*)d_in[100];      // (1024,8)
  const float* bc  = (const float*)d_in[101];      // (8)
  const float* wfw = (const float*)d_in[102];      // (512,8)
  const float* bfw = (const float*)d_in[103];      // (8)

  // ---- workspace carve (bump allocator, ~350 MB total) ----
  char* base = (char*)d_ws;
  size_t off = 0;
  auto alloc = [&](size_t bytes) -> void* {
    void* p = base + off;
    off += (bytes + 255) & ~(size_t)255;
    return p;
  };

  __bf16* wbfC[3][6];
  __bf16* wbfF[3][6];
  for (int l = 0; l < 3; ++l) for (int j = 0; j < 6; ++j) wbfC[l][j] = (__bf16*)alloc(1024L * 1024 * 2);
  for (int l = 0; l < 3; ++l) for (int j = 0; j < 6; ++j) wbfF[l][j] = (__bf16*)alloc(512L * 512 * 2);

  const long RCn = 8L * 1025 * 1024;
  const long LGn = 16L * 1025 * 1025;      // f32 logits elems (max)
  const long SCn = 16L * 1025 * 1032;      // bf16 score elems (max, padded stride)
  float*  xc_f  = (float*) alloc(RCn * 4);
  __bf16* xc_b  = (__bf16*)alloc(RCn * 2);
  __bf16* qb    = (__bf16*)alloc(RCn * 2);
  __bf16* kb    = (__bf16*)alloc(RCn * 2);
  __bf16* vb    = (__bf16*)alloc(RCn * 2);
  float*  logits= (float*) alloc(LGn * 4);
  __bf16* score = (__bf16*)alloc(SCn * 2);
  __bf16* attn  = (__bf16*)alloc(RCn * 2);
  float*  tmp   = (float*) alloc(RCn * 4);
  __bf16* hid   = (__bf16*)alloc(RCn * 2);
  float*  s1row = (float*) alloc(16L * 1025 * 4);
  float*  s2row = (float*) alloc(16L * 1025 * 4);
  int*    idx_c = (int*)   alloc(8L * 128 * 4);
  int*    idx_f = (int*)   alloc(8L * 256 * 4);
  float*  xsc_f = (float*) alloc(8L * 129 * 1024 * 4);
  __bf16* xsc_b = (__bf16*)alloc(8L * 129 * 1024 * 2);
  float*  xf_f  = (float*) alloc(8L * 771 * 512 * 4);
  __bf16* xf_b  = (__bf16*)alloc(8L * 771 * 512 * 2);
  float*  xsf_f = (float*) alloc(8L * 257 * 512 * 4);
  __bf16* xsf_b = (__bf16*)alloc(8L * 257 * 512 * 2);
  float*  relc  = (float*) alloc(64 * 4);
  float*  relf  = (float*) alloc(64 * 4);

  // ---- convert weights to bf16 (every call; deterministic) ----
  const LayerP* CL[3] = {&Lca1, &Lca2, &Lcac};
  const LayerP* FL[3] = {&Lfa1, &Lfa2, &Lfac};
  for (int l = 0; l < 3; ++l) {
    const float* w6c[6] = {CL[l]->wq, CL[l]->wk, CL[l]->wv, CL[l]->wo, CL[l]->w1, CL[l]->w2};
    for (int j = 0; j < 6; ++j) {
      long n = 1024L * 1024;
      cvt_f32_to_bf16_kernel<<<dim3((unsigned)((n + 255) / 256)), dim3(256), 0, stream>>>(w6c[j], wbfC[l][j], n);
    }
    const float* w6f[6] = {FL[l]->wq, FL[l]->wk, FL[l]->wv, FL[l]->wo, FL[l]->w1, FL[l]->w2};
    for (int j = 0; j < 6; ++j) {
      long n = 512L * 512;
      cvt_f32_to_bf16_kernel<<<dim3((unsigned)((n + 255) / 256)), dim3(256), 0, stream>>>(w6f[j], wbfF[l][j], n);
    }
  }

  float* out   = (float*)d_out;
  float* cfeat = out + 64;                       // (8,128,1024)
  float* ffeat = out + 64 + 8L * 128 * 1024;     // (8,256,512)

  // ---- coarse branch ----
  {
    long n = 8L * 1025 * 1024;
    build_xc_kernel<<<dim3((unsigned)((n + 255) / 256)), dim3(256), 0, stream>>>(c_tokens, cls_c, xc_f, xc_b);
  }
  run_layer(stream, 1025, 1024, xc_f, xc_b, Lca1, wbfC[0], qb, kb, vb, logits, score, attn, tmp, hid, s1row);
  run_layer(stream, 1025, 1024, xc_f, xc_b, Lca2, wbfC[1], qb, kb, vb, logits, score, attn, tmp, hid, s2row);
  topk_kernel<<<dim3(8), dim3(256), 0, stream>>>(s1row, s2row, 1025, 1024, 128, idx_c);
  {
    long n = 8L * 129 * 1024;
    gather_c_kernel<<<dim3((unsigned)((n + 255) / 256)), dim3(256), 0, stream>>>(xc_f, idx_c, xsc_f, xsc_b, cfeat);
  }
  run_layer(stream, 129, 1024, xsc_f, xsc_b, Lcac, wbfC[2], qb, kb, vb, logits, score, attn, tmp, hid, s1row);
  rel_kernel<<<dim3(8), dim3(256), 0, stream>>>(xsc_f, 129L * 1024, wc, bc, relc, 1024);

  // ---- fine branch ----
  {
    long n = 8L * 771 * 512;
    build_xf_kernel<<<dim3((unsigned)((n + 255) / 256)), dim3(256), 0, stream>>>(f_tokens, cls_f, idx_c, xf_f, xf_b);
  }
  run_layer(stream, 771, 512, xf_f, xf_b, Lfa1, wbfF[0], qb, kb, vb, logits, score, attn, tmp, hid, s1row);
  run_layer(stream, 771, 512, xf_f, xf_b, Lfa2, wbfF[1], qb, kb, vb, logits, score, attn, tmp, hid, s2row);
  topk_kernel<<<dim3(8), dim3(256), 0, stream>>>(s1row, s2row, 771, 770, 256, idx_f);
  {
    long n = 8L * 257 * 512;
    gather_f_kernel<<<dim3((unsigned)((n + 255) / 256)), dim3(256), 0, stream>>>(xf_f, idx_f, xsf_f, xsf_b, ffeat);
  }
  run_layer(stream, 257, 512, xsf_f, xsf_b, Lfac, wbfF[2], qb, kb, vb, logits, score, attn, tmp, hid, s1row);
  rel_kernel<<<dim3(8), dim3(256), 0, stream>>>(xsf_f, 257L * 512, wfw, bfw, relf, 512);

  final_max_kernel<<<dim3(1), dim3(64), 0, stream>>>(relc, relf, out);
}
